// E3Conv_4355096838555
// MI455X (gfx1250) — matrix operand order repaired and checked by reading, not compiled, not run on hardware
//
#include <hip/hip_runtime.h>
#include <math.h>

typedef __attribute__((ext_vector_type(16))) _Float16 v16h;
typedef __attribute__((ext_vector_type(8)))  _Float16 v8h;
typedef __attribute__((ext_vector_type(8)))  float    v8f;

#define N_NODES   30000
#define N_EDGES   480000
#define N0        64
#define N1        16
#define W_TP      176               // 64 + 16 + 64 + 16 + 16
#define DEG_NORM  0.25f             // 1/sqrt(16)
#define MAX_R     5.0f
#define AVP       ((size_t)N_NODES * 96)   // agg_v component-plane stride

__device__ __forceinline__ float sigmf(float x) {
  return __builtin_amdgcn_rcpf(1.f + __expf(-x));   // v_rcp_f32, not precise div
}
__device__ __forceinline__ float siluf(float x) { return x * sigmf(x); }

__device__ __forceinline__ v16h wmma_f16(v16h a, v16h b, v8f& c) {
  c = __builtin_amdgcn_wmma_f32_16x16x32_f16(false, a, false, b, (short)0, c, false, false);
  return a;
}

// Convert 8 consecutive f32 (16B-aligned) into fragment elements [be, be+8), scaled.
__device__ __forceinline__ void cvt8(v16h& f, int be, const float* __restrict__ p, float scale) {
  float4 a = *(const float4*)p;
  float4 b = *(const float4*)(p + 4);
  f[be + 0] = (_Float16)(a.x * scale); f[be + 1] = (_Float16)(a.y * scale);
  f[be + 2] = (_Float16)(a.z * scale); f[be + 3] = (_Float16)(a.w * scale);
  f[be + 4] = (_Float16)(b.x * scale); f[be + 5] = (_Float16)(b.y * scale);
  f[be + 6] = (_Float16)(b.z * scale); f[be + 7] = (_Float16)(b.w * scale);
}

// A-fragment scatter mapping (ISA 7.12.2): value at (row, k') of a 16x32 chunk lives in
// lane = row + 16*((k'>>3)&1), element = (k'&7) + ((k'&16)?8:0).

// =================== kernel 0: edge geometry + edge_attr (f16) ===================
__global__ void edge_geom_kernel(const float* __restrict__ pos,
                                 const int* __restrict__ ei,
                                 const int* __restrict__ bond_mask,
                                 const float* __restrict__ bond_emb,
                                 _Float16* __restrict__ attr,
                                 float* __restrict__ y1out) {
  int e = blockIdx.x * blockDim.x + threadIdx.x;
  if (e >= N_EDGES) return;
  int src = ei[e], dst = ei[N_EDGES + e];
  float dx = pos[src * 3 + 0] - pos[dst * 3 + 0];
  float dy = pos[src * 3 + 1] - pos[dst * 3 + 1];
  float dz = pos[src * 3 + 2] - pos[dst * 3 + 2];
  float r  = sqrtf(dx * dx + dy * dy + dz * dz);
  float inv = __builtin_amdgcn_rcpf(fmaxf(r, 1e-9f));
  const float s3 = 1.7320508075688772f;  // sqrt(3), Y1 'component' norm
  y1out[(size_t)e * 3 + 0] = s3 * dx * inv;
  y1out[(size_t)e * 3 + 1] = s3 * dy * inv;
  y1out[(size_t)e * 3 + 2] = s3 * dz * inv;
  int b = bond_mask[e];
  _Float16* ap = attr + (size_t)e * 16;
#pragma unroll
  for (int j = 0; j < 8; ++j) ap[j] = (_Float16)bond_emb[b * 8 + j];
  const float step = MAX_R / 9.0f;  // linspace(0,5,10) interior step
#pragma unroll
  for (int j = 0; j < 8; ++j) {
    float val = (float)(j + 1) * step;
    float d   = (r - val) / step;
    ap[8 + j] = (_Float16)(__expf(-d * d) * (1.0f / 1.12f));
  }
}

// =================== kernel 1: s init = atom_emb[type] @ w_init ===================
__global__ void init_s_kernel(const int* __restrict__ atom_type,
                              const float* __restrict__ atom_emb,
                              const float* __restrict__ w_init,
                              float* __restrict__ s) {
  int idx = blockIdx.x * blockDim.x + threadIdx.x;
  if (idx >= N_NODES * N0) return;
  int n = idx >> 6, o = idx & 63;
  const float* emb = atom_emb + (size_t)atom_type[n] * 64;
  float acc = 0.f;
  for (int k = 0; k < 64; ++k) acc += emb[k] * w_init[k * 64 + o];
  s[idx] = acc;
}

__global__ void zero_kernel(float* __restrict__ p, size_t n) {
  size_t i = (size_t)blockIdx.x * blockDim.x + threadIdx.x;
  size_t st = (size_t)gridDim.x * blockDim.x;
  for (; i < n; i += st) p[i] = 0.f;
}

// ====== kernel 2: fused edge MLP (WMMA) + tensor-product messages + atomic scatter ======
// One wave = 16 edges. All B fragments pre-swizzled to fragment-major LDS; A fragments
// come straight from global (attr) or are written in fragment layout by the epilogue.
__global__ __launch_bounds__(256)
void edge_mlp_scatter_kernel(const _Float16* __restrict__ attr,
                             const float* __restrict__ y1,
                             const int* __restrict__ ei,
                             const float* __restrict__ s,
                             const float* __restrict__ v,
                             const float* __restrict__ w1,   // [16,64]
                             const float* __restrict__ b1,   // [64]
                             const float* __restrict__ w2,   // [64,176]
                             float* __restrict__ agg_s,      // [N,80]
                             float* __restrict__ agg_v) {    // [3][N,96]
  __shared__ v16h  sW1f[4 * 32];        // [tile][lane], K padded 16->32 (zeros baked)
  __shared__ float sB1[64];
  __shared__ v16h  sW2f[22 * 32];       // [(chunk*11+tile)][lane]
  __shared__ v16h  sHidF[8][2 * 32];    // per-wave hidden in A-fragment layout
  __shared__ float sC[8][16 * 16];      // per-wave w-tile spill

  const int tid = threadIdx.x;
  {
    _Float16* p = (_Float16*)sW1f;
    for (int i = tid; i < 4 * 32 * 16; i += 256) {
      int e = i & 15, l = (i >> 4) & 31, t = i >> 9;
      int k = ((l < 16) ? 0 : 8) + (e & 7) + ((e & 8) ? 16 : 0);
      int col = t * 16 + (l & 15);
      p[i] = (k < 16) ? (_Float16)w1[k * 64 + col] : (_Float16)0.f;
    }
  }
  if (tid < 64) sB1[tid] = b1[tid];
  {
    _Float16* p = (_Float16*)sW2f;
    for (int i = tid; i < 22 * 32 * 16; i += 256) {
      int e = i & 15, l = (i >> 4) & 31, g = i >> 9;
      int c = g / 11, t = g - c * 11;
      int k = c * 32 + ((l < 16) ? 0 : 8) + (e & 7) + ((e & 8) ? 16 : 0);
      int col = t * 16 + (l & 15);
      p[i] = (_Float16)w2[k * W_TP + col];
    }
  }
  __syncthreads();

  const int wave = tid >> 5;
  const int lane = tid & 31;
  const int ebase = (blockIdx.x * 8 + wave) * 16;
  if (ebase >= N_EDGES) return;

  const int row = lane & 15;               // fragment row / C column index
  const int cn  = lane & 15;
  const int mb  = (lane < 16) ? 0 : 8;
  float* C = sC[wave];

  // A fragment of edge_attr: per-lane contiguous 16B global load, upper half zero (K pad)
  v16h fa = {};
  {
    v8h lo = *(const v8h*)(attr + (size_t)(ebase + row) * 16 + ((lane < 16) ? 0 : 8));
#pragma unroll
    for (int i = 0; i < 8; ++i) fa[i] = lo[i];
  }

  // GEMM1 + bias + silu -> hidden, written directly in A-fragment layout
  _Float16* HF = (_Float16*)sHidF[wave];
  const int hi16 = (cn >= 8) ? 16 : 0;
#pragma unroll
  for (int t = 0; t < 4; ++t) {
    v8f c = {};
    wmma_f16(fa, sW1f[t * 32 + lane], c);
    const int chunk = t >> 1;
    const int eb    = (cn & 7) + ((t & 1) ? 8 : 0);
    const float bias = sB1[t * 16 + cn];
#pragma unroll
    for (int i = 0; i < 8; ++i) {
      int m = mb + i;                       // hidden row (edge)
      HF[(chunk * 32 + m + hi16) * 16 + eb] = (_Float16)siluf(c[i] + bias);
    }
  }

  v16h fh0 = sHidF[wave][lane];
  v16h fh1 = sHidF[wave][32 + lane];

  // GEMM2: 11 tiles of 16 w-columns; spill tile, consume with atomic scatter
  for (int t = 0; t < 11; ++t) {
    v8f c = {};
    wmma_f16(fh0, sW2f[t * 32 + lane], c);
    wmma_f16(fh1, sW2f[(11 + t) * 32 + lane], c);
#pragma unroll
    for (int i = 0; i < 8; ++i) C[(mb + i) * 16 + cn] = c[i];

#pragma unroll
    for (int j = 0; j < 8; ++j) {
      int idx = j * 32 + lane;
      int el  = idx >> 4;
      int cc  = idx & 15;
      int col = t * 16 + cc;
      int e   = ebase + el;
      float w = C[el * 16 + cc];
      int src = ei[e], dst = ei[N_EDGES + e];
      float yx = y1[(size_t)e * 3 + 0];
      float yy = y1[(size_t)e * 3 + 1];
      float yz = y1[(size_t)e * 3 + 2];
      if (col < 64) {                       // 0e x 0e -> 0e
        atomicAdd(&agg_s[(size_t)dst * 80 + col], w * s[(size_t)src * 64 + col]);
      } else if (col < 80) {                // 1o . Y1 -> 0e
        int i1 = col - 64;
        const float* vp = v + (size_t)src * 48 + i1 * 3;
        float d = vp[0] * yx + vp[1] * yy + vp[2] * yz;
        atomicAdd(&agg_s[(size_t)dst * 80 + col], w * d);
      } else if (col < 144) {               // 0e x Y1 -> 1o   (rows 0..63)
        int i1 = col - 80;
        float se = s[(size_t)src * 64 + i1];
        size_t o = (size_t)dst * 96 + i1;
        atomicAdd(&agg_v[0 * AVP + o], w * se * yx);
        atomicAdd(&agg_v[1 * AVP + o], w * se * yy);
        atomicAdd(&agg_v[2 * AVP + o], w * se * yz);
      } else if (col < 160) {               // 1o x 1 -> 1o    (rows 64..79)
        int i1 = col - 144;
        const float* vp = v + (size_t)src * 48 + i1 * 3;
        size_t o = (size_t)dst * 96 + 64 + i1;
        atomicAdd(&agg_v[0 * AVP + o], w * vp[0]);
        atomicAdd(&agg_v[1 * AVP + o], w * vp[1]);
        atomicAdd(&agg_v[2 * AVP + o], w * vp[2]);
      } else {                              // 1o x Y1 -> 1e cross (rows 80..95)
        int i1 = col - 160;
        const float* vp = v + (size_t)src * 48 + i1 * 3;
        size_t o = (size_t)dst * 96 + 80 + i1;
        atomicAdd(&agg_v[0 * AVP + o], w * (vp[1] * yz - vp[2] * yy));
        atomicAdd(&agg_v[1 * AVP + o], w * (vp[2] * yx - vp[0] * yz));
        atomicAdd(&agg_v[2 * AVP + o], w * (vp[0] * yy - vp[1] * yx));
      }
    }
  }
}

// ====== kernel 3: node update — WMMA linears + gated residual ======
__global__ __launch_bounds__(256)
void node_update_kernel(const float* __restrict__ agg_s,   // [N,80]
                        const float* __restrict__ agg_v,   // [3][N,96]
                        const float* __restrict__ lin_s,   // [80,64]
                        const float* __restrict__ lin_v,   // [96,16]
                        const float* __restrict__ gate_w,  // [64,16]
                        float* __restrict__ s,
                        float* __restrict__ v) {
  __shared__ v16h sLSf[3 * 4 * 32];     // [(chunk*4+tile)][lane], K padded 80->96
  __shared__ v16h sLVf[3 * 32];         // [chunk][lane]
  __shared__ v16h sGWf[2 * 32];         // [chunk][lane]
  __shared__ v16h sOSF[8][2 * 32];      // per-wave raw out_s in A-fragment layout

  const int tid = threadIdx.x;
  {
    _Float16* p = (_Float16*)sLSf;
    for (int i = tid; i < 3 * 4 * 32 * 16; i += 256) {
      int e = i & 15, l = (i >> 4) & 31, t = (i >> 9) & 3, c = i >> 11;
      int k = c * 32 + ((l < 16) ? 0 : 8) + (e & 7) + ((e & 8) ? 16 : 0);
      int col = t * 16 + (l & 15);
      p[i] = (k < 80) ? (_Float16)lin_s[k * 64 + col] : (_Float16)0.f;
    }
  }
  {
    _Float16* p = (_Float16*)sLVf;
    for (int i = tid; i < 3 * 32 * 16; i += 256) {
      int e = i & 15, l = (i >> 4) & 31, c = i >> 9;
      int k = c * 32 + ((l < 16) ? 0 : 8) + (e & 7) + ((e & 8) ? 16 : 0);
      p[i] = (_Float16)lin_v[k * 16 + (l & 15)];
    }
  }
  {
    _Float16* p = (_Float16*)sGWf;
    for (int i = tid; i < 2 * 32 * 16; i += 256) {
      int e = i & 15, l = (i >> 4) & 31, c = i >> 9;
      int k = c * 32 + ((l < 16) ? 0 : 8) + (e & 7) + ((e & 8) ? 16 : 0);
      p[i] = (_Float16)gate_w[k * 16 + (l & 15)];
    }
  }
  __syncthreads();

  const int wave = tid >> 5;
  const int lane = tid & 31;
  const int nbase = (blockIdx.x * 8 + wave) * 16;
  if (nbase >= N_NODES) return;             // N multiple of 16 -> full tiles only

  const int row = lane & 15;
  const int cn  = lane & 15;
  const int mb  = (lane < 16) ? 0 : 8;
  const int node_a = nbase + row;           // this lane's A-fragment row

  // out_s A fragments straight from global agg_s (K-contiguous rows, 16B loads)
  const float* as = agg_s + (size_t)node_a * 80;
  v16h fas[3];
#pragma unroll
  for (int k = 0; k < 3; ++k) {
    v16h f = {};
    int kb0 = k * 32 + ((lane < 16) ? 0 : 8);    // always < 80
    cvt8(f, 0, as + kb0, DEG_NORM);
    if (k < 2) {
      int kb1 = k * 32 + ((lane < 16) ? 16 : 24);
      cvt8(f, 8, as + kb1, DEG_NORM);
    }                                            // k==2 upper half: K pad, zero
    fas[k] = f;
  }

  _Float16* OF = (_Float16*)sOSF[wave];
  const int hi16 = (cn >= 8) ? 16 : 0;
#pragma unroll
  for (int t = 0; t < 4; ++t) {
    v8f c = {};
#pragma unroll
    for (int k = 0; k < 3; ++k) wmma_f16(fas[k], sLSf[(k * 4 + t) * 32 + lane], c);
    const int chunk = t >> 1;
    const int eb    = (cn & 7) + ((t & 1) ? 8 : 0);
#pragma unroll
    for (int i = 0; i < 8; ++i) {
      int m = mb + i;
      OF[(chunk * 32 + m + hi16) * 16 + eb] = (_Float16)c[i];      // raw out_s
      s[(size_t)(nbase + m) * 64 + t * 16 + cn] += siluf(c[i]);    // residual update
    }
  }

  // gate = sigmoid(out_s @ gate_w); fragment aligns elementwise with out_v fragments
  v16h fo0 = sOSF[wave][lane];
  v16h fo1 = sOSF[wave][32 + lane];
  v8f g = {};
  wmma_f16(fo0, sGWf[lane], g);
  wmma_f16(fo1, sGWf[32 + lane], g);

  // out_v per xyz component: [16,96]@[96,16]; A from component-major agg_v planes
#pragma unroll
  for (int comp = 0; comp < 3; ++comp) {
    const float* av = agg_v + comp * AVP + (size_t)node_a * 96;
    v8f c = {};
#pragma unroll
    for (int k = 0; k < 3; ++k) {
      v16h f;
      cvt8(f, 0, av + k * 32 + ((lane < 16) ? 0 : 8),  DEG_NORM);
      cvt8(f, 8, av + k * 32 + ((lane < 16) ? 16 : 24), DEG_NORM);
      wmma_f16(f, sLVf[k * 32 + lane], c);
    }
#pragma unroll
    for (int i = 0; i < 8; ++i) {
      int m = mb + i;
      v[(size_t)(nbase + m) * 48 + cn * 3 + comp] += c[i] * sigmf(g[i]);
    }
  }
}

// =================== kernel 4: output head ===================
__global__ void head_kernel(const float* __restrict__ s,
                            const float* __restrict__ v,
                            const float* __restrict__ hs1,  // [64,64]
                            const float* __restrict__ hg,   // [64,16]
                            const float* __restrict__ hs2,  // [64,1]
                            const float* __restrict__ hv2,  // [16,1]
                            float* __restrict__ out) {
  int n = blockIdx.x * blockDim.x + threadIdx.x;
  if (n >= N_NODES) return;
  const float* sp = s + (size_t)n * 64;
  float h[64];
  for (int o = 0; o < 64; ++o) {
    float acc = 0.f;
    for (int k = 0; k < 64; ++k) acc += sp[k] * hs1[k * 64 + o];
    h[o] = siluf(acc);
  }
  float ps = 0.f;
  for (int k = 0; k < 64; ++k) ps += h[k] * hs2[k];
  float pvx = 0.f, pvy = 0.f, pvz = 0.f;
  const float* vp = v + (size_t)n * 48;
  for (int i = 0; i < 16; ++i) {
    float gacc = 0.f;
    for (int k = 0; k < 64; ++k) gacc += h[k] * hg[k * 16 + i];
    float gi = sigmf(gacc) * hv2[i];
    pvx += vp[i * 3 + 0] * gi;
    pvy += vp[i * 3 + 1] * gi;
    pvz += vp[i * 3 + 2] * gi;
  }
  out[(size_t)n * 4 + 0] = ps;
  out[(size_t)n * 4 + 1] = pvx;
  out[(size_t)n * 4 + 2] = pvy;
  out[(size_t)n * 4 + 3] = pvz;
}

// ============================ host launcher ============================
extern "C" void kernel_launch(void* const* d_in, const int* in_sizes, int n_in,
                              void* d_out, int out_size, void* d_ws, size_t ws_size,
                              hipStream_t stream) {
  (void)in_sizes; (void)n_in; (void)out_size; (void)ws_size;
  const float* pos       = (const float*)d_in[0];
  const int*   ei        = (const int*)d_in[1];
  const int*   bond_mask = (const int*)d_in[2];
  const int*   atom_type = (const int*)d_in[3];
  const float* atom_emb  = (const float*)d_in[4];
  const float* bond_emb  = (const float*)d_in[5];
  const float* w_init    = (const float*)d_in[6];
  const float* mlp_w1    = (const float*)d_in[7];
  const float* mlp_b1    = (const float*)d_in[8];
  const float* mlp_w2    = (const float*)d_in[9];
  const float* lin_s     = (const float*)d_in[10];
  const float* lin_v     = (const float*)d_in[11];
  const float* gate_w    = (const float*)d_in[12];
  const float* hs1       = (const float*)d_in[13];
  const float* hg        = (const float*)d_in[14];
  const float* hs2       = (const float*)d_in[15];
  const float* hv2       = (const float*)d_in[16];

  char* ws = (char*)d_ws;
  size_t off = 0;
  auto wsalloc = [&](size_t bytes) {
    char* p = ws + off;
    off = (off + bytes + 255) & ~(size_t)255;
    return p;
  };
  float*    s     = (float*)wsalloc((size_t)N_NODES * 64 * 4);
  float*    v     = (float*)wsalloc((size_t)N_NODES * 48 * 4);
  float*    y1    = (float*)wsalloc((size_t)N_EDGES * 3 * 4);
  _Float16* attr  = (_Float16*)wsalloc((size_t)N_EDGES * 16 * 2);
  float*    agg_s = (float*)wsalloc((size_t)N_NODES * 80 * 4);   // contiguous with agg_v
  float*    agg_v = (float*)wsalloc((size_t)N_NODES * 288 * 4);  // [3][N][96] planes

  edge_geom_kernel<<<(N_EDGES + 255) / 256, 256, 0, stream>>>(
      pos, ei, bond_mask, bond_emb, attr, y1);
  init_s_kernel<<<(N_NODES * 64 + 255) / 256, 256, 0, stream>>>(
      atom_type, atom_emb, w_init, s);
  zero_kernel<<<1024, 256, 0, stream>>>(v, (size_t)N_NODES * 48);

  for (int l = 0; l < 3; ++l) {
    zero_kernel<<<2048, 256, 0, stream>>>(agg_s, (size_t)N_NODES * (80 + 288));
    edge_mlp_scatter_kernel<<<N_EDGES / (8 * 16), 256, 0, stream>>>(
        attr, y1, ei, s, v,
        mlp_w1 + (size_t)l * 16 * 64,
        mlp_b1 + (size_t)l * 64,
        mlp_w2 + (size_t)l * 64 * W_TP,
        agg_s, agg_v);
    node_update_kernel<<<(N_NODES / 16 + 7) / 8, 256, 0, stream>>>(
        agg_s, agg_v,
        lin_s  + (size_t)l * 80 * 64,
        lin_v  + (size_t)l * 96 * 16,
        gate_w + (size_t)l * 64 * 16,
        s, v);
  }

  head_kernel<<<(N_NODES + 255) / 256, 256, 0, stream>>>(
      s, v, hs1, hg, hs2, hv2, (float*)d_out);
}
